// MultiHeadAttention_5823975653887
// MI455X (gfx1250) — matrix operand order repaired and checked by reading, not compiled, not run on hardware
//
#include <hip/hip_runtime.h>
#include <hip/hip_bf16.h>

// MI455X (gfx1250) multi-head attention, all four matmul stages on
// v_wmma_f32_16x16x32_bf16 (wave32, f32 accum), flash-style online softmax.

typedef __attribute__((ext_vector_type(16))) __bf16 v16bf;
typedef __attribute__((ext_vector_type(8)))  __bf16 v8bf;
typedef __attribute__((ext_vector_type(4)))  __bf16 v4bf;
typedef __attribute__((ext_vector_type(8)))  float  v8f;

static __device__ __forceinline__ v16bf cmb(v8bf lo, v8bf hh) {
  return __builtin_shufflevector(lo, hh, 0,1,2,3,4,5,6,7,8,9,10,11,12,13,14,15);
}
static __device__ __forceinline__ v8bf ldb8(const __bf16* p) {
  return *(const v8bf*)p;
}
static __device__ __forceinline__ v8f wmma_bf16(v16bf a, v16bf b, v8f c) {
  // D = A(16x32) * B(32x16) + C, f32 accumulate
  return __builtin_amdgcn_wmma_f32_16x16x32_bf16(false, a, false, b, (short)0, c, false, false);
}

// One K-step's operands for the 32x64 GEMM tile: 2 A tiles + 4 B tiles.
struct GemmTiles {
  v16bf a0, a1, b[4];
};

static __device__ __forceinline__ GemmTiles load_tiles(
    const __bf16* __restrict__ arow0, const __bf16* __restrict__ arow1,
    const __bf16* __restrict__ brow, int k0) {
  GemmTiles t;
  t.a0 = cmb(ldb8(arow0 + k0), ldb8(arow0 + k0 + 16));
  t.a1 = cmb(ldb8(arow1 + k0), ldb8(arow1 + k0 + 16));
  #pragma unroll
  for (int ni = 0; ni < 4; ++ni) {
    const __bf16* bp = brow + (size_t)ni * 16 * 1024 + k0;
    t.b[ni] = cmb(ldb8(bp), ldb8(bp + 16));
  }
  return t;
}

// ---------------- fp32 -> bf16 conversion (4-wide) ----------------
__global__ void cvt4_kernel(const float* __restrict__ in, __bf16* __restrict__ out, int n4) {
  int i = blockIdx.x * blockDim.x + threadIdx.x;
  if (i < n4) {
    float4 v = ((const float4*)in)[i];
    v4bf o;
    o[0] = (__bf16)v.x; o[1] = (__bf16)v.y; o[2] = (__bf16)v.z; o[3] = (__bf16)v.w;
    ((v4bf*)out)[i] = o;
  }
}

// ---------------- QKV projection GEMM ----------------
// C[m,f] = sum_k X[m,k] * W[f,k] + bias[f];  M=8192, N=3072, K=1024.
// Per wave: 32x64 tile, software-pipelined K loop (next step's 12 loads issue
// while current step's 8 WMMAs execute). Consecutive waves share the same
// 64-row weight stripe so B stays hot in WGP$.
// Q,K written [b][h][s][64] bf16 (row-major); V written transposed [b][h][64][s].
__global__ void __launch_bounds__(256) qkv_gemm_kernel(
    const __bf16* __restrict__ X, const __bf16* __restrict__ W,
    const float* __restrict__ bias,
    __bf16* __restrict__ Qb, __bf16* __restrict__ Kb, __bf16* __restrict__ Vt)
{
  const int wave = (int)((blockIdx.x * blockDim.x + threadIdx.x) >> 5);
  const int lane = (int)(threadIdx.x & 31);
  const int lm = lane & 15;
  const int hw = lane >> 4;          // half-wave: 0 or 1
  const int mt = wave % 256;         // 8192 / 32  (fast index -> waves share nt)
  const int nt = wave / 256;         // 3072 / 64
  const int m0 = mt * 32;
  const int n0 = nt * 64;

  const __bf16* arow0 = X + (size_t)(m0 + lm) * 1024 + hw * 8;
  const __bf16* arow1 = arow0 + (size_t)16 * 1024;
  const __bf16* brow  = W + (size_t)(n0 + lm) * 1024 + hw * 8;  // +ni*16*1024

  v8f acc[2][4];
  #pragma unroll
  for (int ni = 0; ni < 4; ++ni) {
    float bv = bias[n0 + ni * 16 + lm];   // lane owns one output feature column
    #pragma unroll
    for (int e = 0; e < 8; ++e) { acc[0][ni][e] = bv; acc[1][ni][e] = bv; }
  }

  GemmTiles cur = load_tiles(arow0, arow1, brow, 0);
  for (int k0 = 0; k0 < 1024 - 32; k0 += 32) {
    GemmTiles nxt = load_tiles(arow0, arow1, brow, k0 + 32);
    #pragma unroll
    for (int ni = 0; ni < 4; ++ni) {
      acc[0][ni] = wmma_bf16(cur.a0, cur.b[ni], acc[0][ni]);
      acc[1][ni] = wmma_bf16(cur.a1, cur.b[ni], acc[1][ni]);
    }
    cur = nxt;
  }
  #pragma unroll
  for (int ni = 0; ni < 4; ++ni) {
    acc[0][ni] = wmma_bf16(cur.a0, cur.b[ni], acc[0][ni]);
    acc[1][ni] = wmma_bf16(cur.a1, cur.b[ni], acc[1][ni]);
  }

  // C layout: lane -> N = lm (fixed), reg r -> M = r + hw*8 (+ mi*16)
  const int bi    = m0 >> 11;        // batch (2048 rows per batch, m0 32-aligned)
  const int sbase = m0 & 2047;
  #pragma unroll
  for (int ni = 0; ni < 4; ++ni) {
    const int f     = n0 + ni * 16 + lm;
    const int which = f >> 10;       // 0=Q 1=K 2=V
    const int fe    = f & 1023;
    const int h     = fe >> 6;
    const int d     = fe & 63;
    #pragma unroll
    for (int mi = 0; mi < 2; ++mi) {
      const int srow = sbase + mi * 16 + hw * 8;
      if (which < 2) {
        __bf16* dst = (which ? Kb : Qb)
                      + ((size_t)(bi * 16 + h) * 2048) * 64 + (size_t)d;
        #pragma unroll
        for (int r = 0; r < 8; ++r)
          dst[(size_t)(srow + r) * 64] = (__bf16)acc[mi][ni][r];
      } else {
        v8bf o;
        #pragma unroll
        for (int r = 0; r < 8; ++r) o[r] = (__bf16)acc[mi][ni][r];
        __bf16* dst = Vt + ((size_t)(bi * 16 + h) * 64 + d) * 2048 + srow;
        *(v8bf*)dst = o;   // 8 consecutive s positions, 16B aligned
      }
    }
  }
}

// ---------------- flash attention ----------------
// One wave per (b,h, 16-query tile). scores^T = K_block(16x64) x Q^T so each
// lane owns one query column; P^T C-tiles map directly onto the A-layout of
// P(16x32) for the PV WMMA (no shuffles). V is pre-transposed so PV B-matrix
// loads are contiguous; V loads are hoisted ahead of the softmax so their
// latency overlaps the exp/reduction chain. Output bf16 [b][s][h*64+d].
__global__ void __launch_bounds__(256) attn_kernel(
    const __bf16* __restrict__ Qb, const __bf16* __restrict__ Kb,
    const __bf16* __restrict__ Vt, __bf16* __restrict__ Ob)
{
  const int wave = (int)((blockIdx.x * blockDim.x + threadIdx.x) >> 5);
  const int lane = (int)(threadIdx.x & 31);
  const int lm = lane & 15;
  const int hw = lane >> 4;
  const int qt = wave & 127;         // 2048/16 query tiles (fast -> same (b,h) per WGP)
  const int bh = wave >> 7;          // [0,64)
  const int b  = bh >> 4, h = bh & 15;
  const int q0 = qt * 16;

  const __bf16* Qp = Qb + (size_t)bh * 2048 * 64;
  const __bf16* Kp = Kb + (size_t)bh * 2048 * 64;
  const __bf16* Vp = Vt + (size_t)bh * 64 * 2048;

  // Q as B-matrix (columns of Q^T): lane lm -> query row q0+lm, K=d contiguous
  const __bf16* qrow = Qp + (size_t)(q0 + lm) * 64 + hw * 8;
  const v16bf qb0 = cmb(ldb8(qrow),      ldb8(qrow + 16));
  const v16bf qb1 = cmb(ldb8(qrow + 32), ldb8(qrow + 48));

  float run_max = -3.4e38f;
  float run_sum = 0.0f;
  v8f acc[4];
  #pragma unroll
  for (int t = 0; t < 4; ++t)
    #pragma unroll
    for (int e = 0; e < 8; ++e) acc[t][e] = 0.0f;

  const __bf16* krow = Kp + (size_t)lm * 64 + hw * 8;     // A-matrix rows = keys
  const __bf16* vrow = Vp + (size_t)lm * 2048 + hw * 8;  // B-matrix cols = Vt rows

  for (int kb = 0; kb < 2048; kb += 32) {
    const __bf16* k0p = krow + (size_t)kb * 64;
    const __bf16* k1p = k0p + 16 * 64;
    v16bf ka00 = cmb(ldb8(k0p),      ldb8(k0p + 16));
    v16bf ka01 = cmb(ldb8(k0p + 32), ldb8(k0p + 48));
    v16bf ka10 = cmb(ldb8(k1p),      ldb8(k1p + 16));
    v16bf ka11 = cmb(ldb8(k1p + 32), ldb8(k1p + 48));

    // V tiles are independent of the score/softmax chain: issue loads now.
    v16bf vB[4];
    #pragma unroll
    for (int t = 0; t < 4; ++t) {
      const __bf16* vp = vrow + (size_t)t * 16 * 2048 + kb;
      vB[t] = cmb(ldb8(vp), ldb8(vp + 16));
    }

    v8f s0, s1;
    #pragma unroll
    for (int e = 0; e < 8; ++e) { s0[e] = 0.0f; s1[e] = 0.0f; }
    s0 = wmma_bf16(ka00, qb0, s0);   // keys kb..kb+15   x queries
    s0 = wmma_bf16(ka01, qb1, s0);
    s1 = wmma_bf16(ka10, qb0, s1);   // keys kb+16..kb+31
    s1 = wmma_bf16(ka11, qb1, s1);

    #pragma unroll
    for (int e = 0; e < 8; ++e) { s0[e] *= 0.125f; s1[e] *= 0.125f; }

    // per-query block max: lane has 16 keys, partner lane (xor 16) other 16
    float m = s0[0];
    #pragma unroll
    for (int e = 1; e < 8; ++e) m = fmaxf(m, s0[e]);
    #pragma unroll
    for (int e = 0; e < 8; ++e) m = fmaxf(m, s1[e]);
    m = fmaxf(m, __shfl_xor(m, 16, 32));

    const float nm   = fmaxf(run_max, m);
    const float corr = __expf(run_max - nm);
    v8f p0, p1;
    float bs = 0.0f;
    #pragma unroll
    for (int e = 0; e < 8; ++e) { p0[e] = __expf(s0[e] - nm); bs += p0[e]; }
    #pragma unroll
    for (int e = 0; e < 8; ++e) { p1[e] = __expf(s1[e] - nm); bs += p1[e]; }
    bs += __shfl_xor(bs, 16, 32);
    run_sum = run_sum * corr + bs;
    run_max = nm;

    // P^T C-tiles -> A-layout of P(16 queries x 32 keys), bf16
    v16bf pA;
    #pragma unroll
    for (int e = 0; e < 8; ++e) { pA[e] = (__bf16)p0[e]; pA[e + 8] = (__bf16)p1[e]; }

    // rescale output accumulators (reg r holds query q0 + r + hw*8)
    #pragma unroll
    for (int r = 0; r < 8; ++r) {
      const float cr = __shfl(corr, r + hw * 8, 32);
      #pragma unroll
      for (int t = 0; t < 4; ++t) acc[t][r] *= cr;
    }

    // PV: 4 d-chunks of 16
    #pragma unroll
    for (int t = 0; t < 4; ++t)
      acc[t] = wmma_bf16(pA, vB[t], acc[t]);
  }

  const float inv = 1.0f / run_sum;
  #pragma unroll
  for (int r = 0; r < 8; ++r) {
    const float ir = __shfl(inv, r + hw * 8, 32);
    const int s = q0 + hw * 8 + r;
    __bf16* orow = Ob + (size_t)(b * 2048 + s) * 1024 + h * 64;
    #pragma unroll
    for (int t = 0; t < 4; ++t)
      orow[t * 16 + lm] = (__bf16)(acc[t][r] * ir);
  }
}

// ---------------- output projection GEMM ----------------
// Y[m,f] = sum_k A[m,k] * W[f,k] + bias[f];  M=8192, N=1024, K=1024, fp32 out.
__global__ void __launch_bounds__(256) proj_gemm_kernel(
    const __bf16* __restrict__ Ain, const __bf16* __restrict__ W,
    const float* __restrict__ bias, float* __restrict__ Y)
{
  const int wave = (int)((blockIdx.x * blockDim.x + threadIdx.x) >> 5);
  const int lane = (int)(threadIdx.x & 31);
  const int lm = lane & 15;
  const int hw = lane >> 4;
  const int mt = wave % 256;         // 8192 / 32  (fast index -> waves share nt)
  const int nt = wave / 256;         // 1024 / 64
  const int m0 = mt * 32;
  const int n0 = nt * 64;

  const __bf16* arow0 = Ain + (size_t)(m0 + lm) * 1024 + hw * 8;
  const __bf16* arow1 = arow0 + (size_t)16 * 1024;
  const __bf16* brow  = W + (size_t)(n0 + lm) * 1024 + hw * 8;

  v8f acc[2][4];
  #pragma unroll
  for (int ni = 0; ni < 4; ++ni) {
    float bv = bias[n0 + ni * 16 + lm];
    #pragma unroll
    for (int e = 0; e < 8; ++e) { acc[0][ni][e] = bv; acc[1][ni][e] = bv; }
  }

  GemmTiles cur = load_tiles(arow0, arow1, brow, 0);
  for (int k0 = 0; k0 < 1024 - 32; k0 += 32) {
    GemmTiles nxt = load_tiles(arow0, arow1, brow, k0 + 32);
    #pragma unroll
    for (int ni = 0; ni < 4; ++ni) {
      acc[0][ni] = wmma_bf16(cur.a0, cur.b[ni], acc[0][ni]);
      acc[1][ni] = wmma_bf16(cur.a1, cur.b[ni], acc[1][ni]);
    }
    cur = nxt;
  }
  #pragma unroll
  for (int ni = 0; ni < 4; ++ni) {
    acc[0][ni] = wmma_bf16(cur.a0, cur.b[ni], acc[0][ni]);
    acc[1][ni] = wmma_bf16(cur.a1, cur.b[ni], acc[1][ni]);
  }

  #pragma unroll
  for (int ni = 0; ni < 4; ++ni) {
    const int n = n0 + ni * 16 + lm;
    #pragma unroll
    for (int mi = 0; mi < 2; ++mi) {
      const int mrow = m0 + mi * 16 + hw * 8;
      #pragma unroll
      for (int r = 0; r < 8; ++r)
        Y[(size_t)(mrow + r) * 1024 + n] = acc[mi][ni][r];
    }
  }
}

extern "C" void kernel_launch(void* const* d_in, const int* in_sizes, int n_in,
                              void* d_out, int out_size, void* d_ws, size_t ws_size,
                              hipStream_t stream) {
  (void)in_sizes; (void)n_in; (void)out_size; (void)ws_size;
  const float* x      = (const float*)d_in[0];   // [4,2048,1024]
  const float* w_qkv  = (const float*)d_in[1];   // [3072,1024]
  const float* b_qkv  = (const float*)d_in[2];   // [3072]
  const float* w_proj = (const float*)d_in[3];   // [1024,1024]
  const float* b_proj = (const float*)d_in[4];   // [1024]
  float* y = (float*)d_out;                      // [4,2048,1024]

  char* ws = (char*)d_ws;
  size_t off = 0;
  auto take = [&](size_t bytes) {
    char* p = ws + off;
    off += (bytes + 255) & ~(size_t)255;
    return p;
  };
  __bf16* xb  = (__bf16*)take((size_t)8192 * 1024 * 2);   // x bf16 (reused as attn out)
  __bf16* wqb = (__bf16*)take((size_t)3072 * 1024 * 2);
  __bf16* wpb = (__bf16*)take((size_t)1024 * 1024 * 2);
  __bf16* dq  = (__bf16*)take((size_t)4 * 16 * 2048 * 64 * 2);  // [b][h][s][d]
  __bf16* dk  = (__bf16*)take((size_t)4 * 16 * 2048 * 64 * 2);  // [b][h][s][d]
  __bf16* dvt = (__bf16*)take((size_t)4 * 16 * 64 * 2048 * 2);  // [b][h][d][s]
  __bf16* dao = xb;  // attention output reuses x-bf16 region (dead after QKV GEMM)

  cvt4_kernel<<<(8192 * 1024 / 4 + 255) / 256, 256, 0, stream>>>(x, xb, 8192 * 1024 / 4);
  cvt4_kernel<<<(3072 * 1024 / 4 + 255) / 256, 256, 0, stream>>>(w_qkv, wqb, 3072 * 1024 / 4);
  cvt4_kernel<<<(1024 * 1024 / 4 + 255) / 256, 256, 0, stream>>>(w_proj, wpb, 1024 * 1024 / 4);

  // 256 M-tiles * 48 N-tiles = 12288 waves / 8 per block
  qkv_gemm_kernel<<<12288 / 8, 256, 0, stream>>>(xb, wqb, b_qkv, dq, dk, dvt);
  // 64 (b,h) * 128 q-tiles = 8192 waves
  attn_kernel<<<8192 / 8, 256, 0, stream>>>(dq, dk, dvt, dao);
  // 256 M-tiles * 16 N-tiles = 4096 waves
  proj_gemm_kernel<<<4096 / 8, 256, 0, stream>>>(dao, wpb, b_proj, y);
}
